// BasedAttention_15891378995553
// MI455X (gfx1250) — matrix operand order repaired and checked by reading, not compiled, not run on hardware
//
#include <hip/hip_runtime.h>

typedef __bf16 bf16_t;
typedef __attribute__((ext_vector_type(16))) __bf16 v16bf;
typedef __attribute__((ext_vector_type(8)))  float  v8f;

union Frag { v16bf v; uint4 u[2]; };

__device__ __forceinline__ v8f wmma_bf16(v16bf a, v16bf b, v8f c) {
  return __builtin_amdgcn_wmma_f32_16x16x32_bf16(false, a, false, b, (short)0, c, false, false);
}

__device__ __forceinline__ int clampi(int v, int lo, int hi) {
  return v < lo ? lo : (v > hi ? hi : v);
}

#define N_SEQ 1024
#define D_IN  1024
#define NOUT  1536   // 256 Q + 256 K + 1024 V
#define DTP   288    // DT=273 padded to multiple of 16
#define SQRT_HALF 0.70710678118654752440f

// ---- CDNA5 async global->LDS path (builtin exists: probe round 2) ----
#if __has_builtin(__builtin_amdgcn_global_load_async_to_lds_b128)
#define HAVE_ASYNC_LDS 1
#else
#define HAVE_ASYNC_LDS 0
#endif

typedef int v4i_vs __attribute__((vector_size(16)));
#define GAS1(p) ((__attribute__((address_space(1))) v4i_vs*)(p))
#define LAS3(p) ((__attribute__((address_space(3))) v4i_vs*)(p))

__device__ __forceinline__ void wait_async_zero() {
#if HAVE_ASYNC_LDS
#if __has_builtin(__builtin_amdgcn_s_wait_asynccnt)
  __builtin_amdgcn_s_wait_asynccnt(0);
#else
  asm volatile("s_wait_asynccnt 0x0" ::: "memory");
#endif
#endif
}

// ---------------- conversion kernels ----------------

__global__ __launch_bounds__(256) void cvt_x_kernel(const float* __restrict__ x,
                                                    bf16_t* __restrict__ xb, int n) {
  int i = blockIdx.x * 256 + threadIdx.x;
  if (i < n) xb[i] = (bf16_t)x[i];
}

__global__ __launch_bounds__(256) void build_w_kernel(
    const float* __restrict__ Wq, const float* __restrict__ bq,
    const float* __restrict__ Wk, const float* __restrict__ bk,
    const float* __restrict__ Wv, const float* __restrict__ bv,
    bf16_t* __restrict__ wcat, float* __restrict__ bcat) {
  int i = blockIdx.x * 256 + threadIdx.x;   // over 1536*1024
  if (i >= NOUT * D_IN) return;
  int f = i / D_IN, d = i % D_IN;
  float val;
  if (f < 256)       val = Wq[f * D_IN + d];
  else if (f < 512)  val = Wk[(f - 256) * D_IN + d];
  else               val = Wv[(f - 512) * D_IN + d];
  wcat[i] = (bf16_t)val;
  if (d == 0) {
    float bvv;
    if (f < 256)       bvv = bq[f];
    else if (f < 512)  bvv = bk[f - 256];
    else               bvv = bv[f - 512];
    bcat[f] = bvv;
  }
}

// ---------------- fused QKV GEMM ----------------
// One wave per 32x32 output tile (2x2 grid of 16x16 WMMA tiles, shared fragments).
// Q,K (cols<512) -> O as f32; V (cols>=512) -> Vbf as bf16.

__global__ __launch_bounds__(256) void qkv_gemm_kernel(
    const bf16_t* __restrict__ xb, const bf16_t* __restrict__ wmat,
    const float* __restrict__ bias, float* __restrict__ O,
    bf16_t* __restrict__ Vbf) {
  int lane = threadIdx.x & 31;
  int wave = threadIdx.x >> 5;
  int tile = blockIdx.x * 8 + wave;       // 1536 tiles total (32 x 48)
  int tr = tile / 48;                     // sequence tile (32 rows)
  int tc = tile % 48;                     // feature tile (32 cols)
  int lr = lane & 15;
  int hi = lane >> 4;
  int aoff = hi ? 8 : 0;
  const bf16_t* arow0 = xb + (size_t)(tr * 32 + lr) * D_IN;
  const bf16_t* arow1 = arow0 + 16 * D_IN;
  const bf16_t* brow0 = wmat + (size_t)(tc * 32 + lr) * D_IN;
  const bf16_t* brow1 = brow0 + 16 * D_IN;
  v8f acc00 = {0.f,0.f,0.f,0.f,0.f,0.f,0.f,0.f};
  v8f acc01 = acc00, acc10 = acc00, acc11 = acc00;
  for (int k = 0; k < D_IN; k += 32) {
    __builtin_prefetch(arow0 + k + 256, 0, 3);
    __builtin_prefetch(brow0 + k + 256, 0, 3);
    Frag a0, a1, b0, b1;
    a0.u[0] = *(const uint4*)(arow0 + k + aoff);
    a0.u[1] = *(const uint4*)(arow0 + k + aoff + 16);
    a1.u[0] = *(const uint4*)(arow1 + k + aoff);
    a1.u[1] = *(const uint4*)(arow1 + k + aoff + 16);
    b0.u[0] = *(const uint4*)(brow0 + k + aoff);
    b0.u[1] = *(const uint4*)(brow0 + k + aoff + 16);
    b1.u[0] = *(const uint4*)(brow1 + k + aoff);
    b1.u[1] = *(const uint4*)(brow1 + k + aoff + 16);
    acc00 = wmma_bf16(a0.v, b0.v, acc00);
    acc01 = wmma_bf16(a0.v, b1.v, acc01);
    acc10 = wmma_bf16(a1.v, b0.v, acc10);
    acc11 = wmma_bf16(a1.v, b1.v, acc11);
  }
#pragma unroll
  for (int t = 0; t < 4; ++t) {
    v8f acc = (t == 0) ? acc00 : (t == 1) ? acc01 : (t == 2) ? acc10 : acc11;
    int rb = tr * 32 + (t >> 1) * 16 + 8 * hi;
    int col = tc * 32 + (t & 1) * 16 + lr;
    float bvv = bias[col];
#pragma unroll
    for (int v = 0; v < 8; ++v) {
      float val = acc[v] + bvv;
      if (col < 512) O[(size_t)(rb + v) * NOUT + col] = val;
      else           Vbf[(size_t)(rb + v) * 1024 + col - 512] = (bf16_t)val;
    }
  }
}

// ---------------- sliding-window softmax attention (writes out) ----------------
// One wave per (head, 16-query block). Keys covered: [n0-64, n0+32) = 96 positions.

__global__ __launch_bounds__(128) void window_attn_kernel(
    const float* __restrict__ O, const bf16_t* __restrict__ Vbf,
    float* __restrict__ out) {
  __shared__ float  sc[4][16 * 96];
  __shared__ bf16_t wt[4][16 * 96];
  int lane = threadIdx.x & 31;
  int w = threadIdx.x >> 5;
  int qb = blockIdx.x * 4 + w;           // 1024 blocks total
  int h = qb >> 6;
  int n0 = (qb & 63) * 16;
  int mstart = n0 - 64;
  int lr = lane & 15;
  int hi = lane >> 4;
  int aoff = hi ? 8 : 0;

  // A fragment: Q rows (16 x 16dp, K padded to 32 with zeros)
  v16bf aq;
  const float* Qrow = O + (size_t)(n0 + lr) * NOUT + h * 16;
#pragma unroll
  for (int e = 0; e < 8; ++e) {
    aq[e] = (bf16_t)Qrow[aoff + e];
    aq[8 + e] = (bf16_t)0.0f;
  }
  // scores: 6 key tiles of 16
  for (int kt = 0; kt < 6; ++kt) {
    int m = mstart + kt * 16 + lr;
    int mc = clampi(m, 0, N_SEQ - 1);
    const float* Krow = O + (size_t)mc * NOUT + 256 + h * 16;
    v16bf bfrag;
#pragma unroll
    for (int e = 0; e < 16; ++e)
      bfrag[e] = (e < 8) ? (bf16_t)Krow[aoff + e] : (bf16_t)0.0f;
    v8f g = {0.f,0.f,0.f,0.f,0.f,0.f,0.f,0.f};
    g = wmma_bf16(aq, bfrag, g);
#pragma unroll
    for (int v = 0; v < 8; ++v)
      sc[w][(v + 8 * hi) * 96 + kt * 16 + lr] = g[v];
  }
  __syncthreads();
  // softmax per row (lanes 0..15 each own one query row)
  if (lane < 16) {
    int jlo = lane + 1;                  // m >= n-63
    int jclip = 64 - n0;                 // m >= 0
    if (jclip > jlo) jlo = jclip;
    int jhi = lane + 64;                 // m <= n
    float mx = -1e30f;
    for (int j = jlo; j <= jhi; ++j) mx = fmaxf(mx, sc[w][lane * 96 + j]);
    float sum = 0.f;
    for (int j = 0; j < 96; ++j) {
      float e = 0.f;
      if (j >= jlo && j <= jhi) e = __expf(sc[w][lane * 96 + j] - mx);
      sc[w][lane * 96 + j] = e;
      sum += e;
    }
    float inv = 1.0f / sum;
    for (int j = 0; j < 96; ++j) wt[w][lane * 96 + j] = (bf16_t)(sc[w][lane * 96 + j] * inv);
  }
  __syncthreads();
  // y_loc = wts(16x96) x V_win(96x64)
  for (int dj = 0; dj < 4; ++dj) {
    v8f acc = {0.f,0.f,0.f,0.f,0.f,0.f,0.f,0.f};
    int d = dj * 16 + lr;
    for (int kk = 0; kk < 96; kk += 32) {
      Frag a;
      const bf16_t* wrow = &wt[w][lr * 96 + kk + aoff];
      a.u[0] = *(const uint4*)(wrow);
      a.u[1] = *(const uint4*)(wrow + 16);
      v16bf bfrag;
#pragma unroll
      for (int e = 0; e < 16; ++e) {
        int m = mstart + kk + aoff + e + (e >= 8 ? 8 : 0);
        int mc = clampi(m, 0, N_SEQ - 1);
        bfrag[e] = Vbf[(size_t)mc * 1024 + h * 64 + d];
      }
      acc = wmma_bf16(a.v, bfrag, acc);
    }
#pragma unroll
    for (int v = 0; v < 8; ++v)
      out[(size_t)(n0 + v + 8 * hi) * 1024 + h * 64 + d] = acc[v];
  }
}

// ---------------- chunked linear attention (adds into out) ----------------
// One workgroup (8 waves) per head; 16 sequential chunks of 64.
// Next chunk's Q/K/V staged to LDS with GLOBAL_LOAD_ASYNC_TO_LDS (double-buffered).

__global__ __launch_bounds__(256) void linear_attn_kernel(
    const float* __restrict__ O, const bf16_t* __restrict__ Vbf,
    float* __restrict__ out) {
  extern __shared__ char smem[];
  float*  S    = (float*)smem;               // 288*64
  float*  Qf   = S + DTP * 64;               // 2 x 64*16 (double buffered)
  float*  Kf   = Qf + 2 * 64 * 16;           // 2 x 64*16
  float*  numb = Kf + 2 * 64 * 16;           // 64*64
  float*  z    = numb + 64 * 64;             // 288
  float*  den  = z + DTP;                    // 64
  bf16_t* phiq = (bf16_t*)(den + 64);        // 64*288
  bf16_t* phik = phiq + 64 * DTP;            // 64*288
  bf16_t* P    = phik + 64 * DTP;            // 64*64
  bf16_t* Vc   = P + 64 * 64;                // 2 x 64*64 (double buffered)
  bf16_t* Qc   = Vc + 2 * 64 * 64;           // 64*32 (K padded)
  bf16_t* Kc   = Qc + 64 * 32;               // 64*32

  int tid = threadIdx.x;
  int lane = tid & 31;
  int w = tid >> 5;
  int lr = lane & 15;
  int hi = lane >> 4;
  int aoff = hi ? 8 : 0;
  int h = blockIdx.x;

  auto issue_loads = [&](int cc, int buf) {
    float*  QfB = Qf + buf * 1024;
    float*  KfB = Kf + buf * 1024;
    bf16_t* VcB = Vc + buf * 4096;
    int nb = cc * 64;
#if HAVE_ASYNC_LDS
    {
      int row = tid >> 2, c4 = (tid & 3) * 4;   // 64 rows x 4 16B chunks of Q and K
      __builtin_amdgcn_global_load_async_to_lds_b128(
          GAS1(O + (size_t)(nb + row) * NOUT + h * 16 + c4),
          LAS3(QfB + row * 16 + c4), 0, 0);
      __builtin_amdgcn_global_load_async_to_lds_b128(
          GAS1(O + (size_t)(nb + row) * NOUT + 256 + h * 16 + c4),
          LAS3(KfB + row * 16 + c4), 0, 0);
      for (int i = tid; i < 512; i += 256) {    // 64 rows x 8 16B chunks of V (bf16)
        int vr = i >> 3, vc8 = (i & 7) * 8;
        __builtin_amdgcn_global_load_async_to_lds_b128(
            GAS1(Vbf + (size_t)(nb + vr) * 1024 + h * 64 + vc8),
            LAS3(VcB + vr * 64 + vc8), 0, 0);
      }
    }
#else
    for (int i = tid; i < 64 * 16; i += 256) {
      int n = i >> 4, q = i & 15;
      QfB[i] = O[(size_t)(nb + n) * NOUT + h * 16 + q];
      KfB[i] = O[(size_t)(nb + n) * NOUT + 256 + h * 16 + q];
    }
    for (int i = tid; i < 64 * 64; i += 256)
      VcB[i] = Vbf[(size_t)(nb + (i >> 6)) * 1024 + h * 64 + (i & 63)];
#endif
  };

  for (int i = tid; i < DTP * 64; i += 256) S[i] = 0.f;
  for (int i = tid; i < DTP; i += 256) z[i] = 0.f;
#if HAVE_ASYNC_LDS
  issue_loads(0, 0);
  wait_async_zero();
#endif
  __syncthreads();

  for (int c0 = 0; c0 < 16; ++c0) {
    int cur = c0 & 1;
    int nbase = c0 * 64;
    float*  QfC = Qf + cur * 1024;
    float*  KfC = Kf + cur * 1024;
    bf16_t* VcC = Vc + cur * 4096;
#if HAVE_ASYNC_LDS
    if (c0 < 15) issue_loads(c0 + 1, cur ^ 1);  // overlap next chunk with compute
#else
    issue_loads(c0, cur);
    __syncthreads();
#endif
    // Phase A2: build padded bf16 Q/K tiles + phi features
    for (int i = tid; i < 64 * 16; i += 256) {
      int n = i >> 4, q = i & 15;
      Qc[n * 32 + q] = (bf16_t)QfC[i]; Qc[n * 32 + 16 + q] = (bf16_t)0.f;
      Kc[n * 32 + q] = (bf16_t)KfC[i]; Kc[n * 32 + 16 + q] = (bf16_t)0.f;
    }
    for (int i = tid; i < 64 * DTP; i += 256) {
      int n = i / DTP, t = i % DTP;
      float vq, vk;
      if (t == 0)       { vq = 1.f; vk = 1.f; }
      else if (t < 17)  { vq = QfC[n * 16 + t - 1]; vk = KfC[n * 16 + t - 1]; }
      else if (t < 273) {
        int ij = t - 17; int ii = ij >> 4; int jj = ij & 15;
        vq = QfC[n * 16 + ii] * QfC[n * 16 + jj] * SQRT_HALF;
        vk = KfC[n * 16 + ii] * KfC[n * 16 + jj] * SQRT_HALF;
      } else            { vq = 0.f; vk = 0.f; }
      phiq[i] = (bf16_t)vq;
      phik[i] = (bf16_t)vk;
    }
    __syncthreads();
    // Phase B: G = Qc x Kc^T, P = causal_mask(1 + g + 0.5 g^2)
    for (int t8 = w; t8 < 16; t8 += 8) {
      int ti = t8 >> 2, tj = t8 & 3;
      Frag a, b;
      a.u[0] = *(const uint4*)&Qc[(ti * 16 + lr) * 32 + aoff];
      a.u[1] = *(const uint4*)&Qc[(ti * 16 + lr) * 32 + aoff + 16];
      b.u[0] = *(const uint4*)&Kc[(tj * 16 + lr) * 32 + aoff];
      b.u[1] = *(const uint4*)&Kc[(tj * 16 + lr) * 32 + aoff + 16];
      v8f g = {0.f,0.f,0.f,0.f,0.f,0.f,0.f,0.f};
      g = wmma_bf16(a.v, b.v, g);
#pragma unroll
      for (int v = 0; v < 8; ++v) {
        int qn = ti * 16 + v + 8 * hi;
        int km = tj * 16 + lr;
        float gv = g[v];
        float p = (km <= qn) ? (1.f + gv + 0.5f * gv * gv) : 0.f;
        P[qn * 64 + km] = (bf16_t)p;
      }
    }
    __syncthreads();
    // Phase C: den[n] = sum_m P[n,m] + phiq[n].z_prev + eps
    if (tid < 64) {
      float s = 1e-8f;
      for (int m = 0; m < 64; ++m) s += (float)P[tid * 64 + m];
      for (int t = 0; t < DTP; ++t) s += (float)phiq[tid * DTP + t] * z[t];
      den[tid] = s;
    }
    // Phase D: num = P x Vc + phiq x S_prev
    for (int t8 = w; t8 < 16; t8 += 8) {
      int ti = t8 >> 2, dj = t8 & 3;
      int d = dj * 16 + lr;
      v8f acc = {0.f,0.f,0.f,0.f,0.f,0.f,0.f,0.f};
      for (int kk = 0; kk < 64; kk += 32) {
        Frag a;
        a.u[0] = *(const uint4*)&P[(ti * 16 + lr) * 64 + kk + aoff];
        a.u[1] = *(const uint4*)&P[(ti * 16 + lr) * 64 + kk + aoff + 16];
        v16bf bfrag;
#pragma unroll
        for (int e = 0; e < 16; ++e)
          bfrag[e] = VcC[(kk + aoff + e + (e >= 8 ? 8 : 0)) * 64 + d];
        acc = wmma_bf16(a.v, bfrag, acc);
      }
      for (int kk = 0; kk < DTP; kk += 32) {
        Frag a;
        a.u[0] = *(const uint4*)&phiq[(ti * 16 + lr) * DTP + kk + aoff];
        a.u[1] = *(const uint4*)&phiq[(ti * 16 + lr) * DTP + kk + aoff + 16];
        v16bf bfrag;
#pragma unroll
        for (int e = 0; e < 16; ++e)
          bfrag[e] = (bf16_t)S[(kk + aoff + e + (e >= 8 ? 8 : 0)) * 64 + d];
        acc = wmma_bf16(a.v, bfrag, acc);
      }
#pragma unroll
      for (int v = 0; v < 8; ++v)
        numb[(ti * 16 + v + 8 * hi) * 64 + d] = acc[v];
    }
    __syncthreads();
    // Phase E1: out += num / den
    for (int i = tid; i < 64 * 64; i += 256) {
      int n = i >> 6, d = i & 63;
      out[(size_t)(nbase + n) * 1024 + h * 64 + d] += numb[i] / den[n];
    }
    // Phase E2: z += column sums of phik
    for (int t = tid; t < DTP; t += 256) {
      float acc = z[t];
      for (int n = 0; n < 64; ++n) acc += (float)phik[n * DTP + t];
      z[t] = acc;
    }
    // Phase E3: S += phik^T x Vc (18 x 4 tiles, K=64)
    for (int t8 = w; t8 < 72; t8 += 8) {
      int ti = t8 >> 2, dj = t8 & 3;
      int d = dj * 16 + lr;
      v8f acc;
#pragma unroll
      for (int v = 0; v < 8; ++v) acc[v] = S[(ti * 16 + v + 8 * hi) * 64 + d];
      int trow = ti * 16 + lr;
      for (int kk = 0; kk < 64; kk += 32) {
        v16bf a;
#pragma unroll
        for (int e = 0; e < 8; ++e) {
          a[e]     = phik[(kk + aoff + e) * DTP + trow];
          a[8 + e] = phik[(kk + aoff + 16 + e) * DTP + trow];
        }
        v16bf bfrag;
#pragma unroll
        for (int e = 0; e < 16; ++e)
          bfrag[e] = VcC[(kk + aoff + e + (e >= 8 ? 8 : 0)) * 64 + d];
        acc = wmma_bf16(a, bfrag, acc);
      }
#pragma unroll
      for (int v = 0; v < 8; ++v) S[(ti * 16 + v + 8 * hi) * 64 + d] = acc[v];
    }
    wait_async_zero();   // next chunk's async tiles have landed
    __syncthreads();
  }
}

// ---------------- launcher ----------------

extern "C" void kernel_launch(void* const* d_in, const int* in_sizes, int n_in,
                              void* d_out, int out_size, void* d_ws, size_t ws_size,
                              hipStream_t stream) {
  (void)in_sizes; (void)n_in; (void)out_size; (void)ws_size;
  const float* x  = (const float*)d_in[0];
  const float* Wq = (const float*)d_in[1];
  const float* bq = (const float*)d_in[2];
  const float* Wk = (const float*)d_in[3];
  const float* bk = (const float*)d_in[4];
  const float* Wv = (const float*)d_in[5];
  const float* bv = (const float*)d_in[6];
  float* out = (float*)d_out;
  char* ws = (char*)d_ws;

  bf16_t* xb   = (bf16_t*)(ws);                        // 2 MB
  bf16_t* wcat = (bf16_t*)(ws + 2097152);              // 3 MB
  float*  bcat = (float*) (ws + 5242880);              // 6 KB
  float*  O    = (float*) (ws + 5249024);              // 6 MB (Q,K f32; V slots unused)
  bf16_t* Vbf  = (bf16_t*)(ws + 11540480);             // 2 MB (V in bf16)

  cvt_x_kernel<<<dim3(4096), dim3(256), 0, stream>>>(x, xb, N_SEQ * D_IN);
  build_w_kernel<<<dim3(6144), dim3(256), 0, stream>>>(Wq, bq, Wk, bk, Wv, bv, wcat, bcat);
  qkv_gemm_kernel<<<dim3(192), dim3(256), 0, stream>>>(xb, wcat, bcat, O, Vbf);
  window_attn_kernel<<<dim3(256), dim3(128), 0, stream>>>(O, Vbf, out);

  const int smem_bytes =
      (DTP * 64 + 2 * 64 * 16 * 2 + 64 * 64 + DTP + 64) * 4      // f32 part
    + (2 * 64 * DTP + 64 * 64 + 2 * 64 * 64 + 2 * 64 * 32) * 2;  // bf16 part
  (void)hipFuncSetAttribute((const void*)linear_attn_kernel,
                            hipFuncAttributeMaxDynamicSharedMemorySize, smem_bytes);
  linear_attn_kernel<<<dim3(16), dim3(256), smem_bytes, stream>>>(O, Vbf, out);
}
